// ANN_62148176773160
// MI455X (gfx1250) — matrix operand order repaired and checked by reading, not compiled, not run on hardware
//
#include <hip/hip_runtime.h>
#include <hip/hip_bf16.h>

#define BB    8
#define NUSR  50
#define KCH   20
#define DD    3
#define HH    128
#define FFD   2048
#define LL    3
#define NHD   8
#define SS    1000          // N*K
#define HDD   16            // head dim
#define MM    (BB * SS)     // 8000 rows, divisible by 16
#define SPAD  1024          // S padded to multiple of 32 for WMMA K-steps
#define STILES 63           // ceil(S/16)
#define NTW   8             // n-tiles per wave in the GEMM (16x128 block per wave)

typedef _Float16 half16 __attribute__((ext_vector_type(16)));
typedef float    float8 __attribute__((ext_vector_type(8)));

// ---------------------------------------------------------------------------
// Weight convert + transpose: W[K,N] f32 -> Wt[N,K] f16
// ---------------------------------------------------------------------------
__global__ void wt_convert(const float* __restrict__ W, _Float16* __restrict__ Wt,
                           int Kd, int Ncols) {
  const int idx = blockIdx.x * blockDim.x + threadIdx.x;
  if (idx >= Kd * Ncols) return;
  const int n = idx / Kd;
  const int k = idx % Kd;
  Wt[idx] = (_Float16)W[(size_t)k * Ncols + n];
}

// ---------------------------------------------------------------------------
// x0 = state @ W_pre + b_pre  (D=3 contraction, elementwise kernel)
// ---------------------------------------------------------------------------
__global__ void pre_proj(const float* __restrict__ state, const float* __restrict__ Wp,
                         const float* __restrict__ bp, float* __restrict__ xF,
                         _Float16* __restrict__ xH) {
  const int idx = blockIdx.x * blockDim.x + threadIdx.x;
  if (idx >= MM * HH) return;
  const int h = idx % HH;
  const size_t bs = idx / HH;
  const float* sp = state + bs * DD;
  float x = bp[h];
#pragma unroll
  for (int d = 0; d < DD; ++d) x += sp[d] * Wp[d * HH + h];
  xF[idx] = x;
  xH[idx] = (_Float16)x;
}

// ---------------------------------------------------------------------------
// Mask + action-index kernel (one block per batch row)
// ---------------------------------------------------------------------------
__global__ void mask_kernel(const float* __restrict__ prev_state,
                            const float* __restrict__ state,
                            int* __restrict__ mask, int* __restrict__ actIdx) {
  __shared__ int st[SS];
  __shared__ int asg[SS];
  __shared__ int uany[NUSR];
  __shared__ int full[KCH];
  const int b = blockIdx.x;
  for (int s = threadIdx.x; s < SS; s += blockDim.x) {
    const float* sp = state + ((size_t)(b * SS + s)) * DD;
    const int stat = (sp[DD - 1] != 0.f);
    st[s] = stat;
    float sum = 0.f;
#pragma unroll
    for (int d = 0; d < DD; ++d) sum += sp[d];
    asg[s] = (sum != 0.f);
    const int pstat = (prev_state[((size_t)(b * SS + s)) * DD + DD - 1] != 0.f);
    if (pstat != stat) actIdx[b] = s;   // exactly one change per row
  }
  __syncthreads();
  for (int n = threadIdx.x; n < NUSR; n += blockDim.x) {
    int any = 0;
    for (int k = 0; k < KCH; ++k) any |= st[k * NUSR + n];
    uany[n] = any;
  }
  for (int k = threadIdx.x; k < KCH; k += blockDim.x) {
    int c = 0;
    for (int n = 0; n < NUSR; ++n) c += asg[k * NUSR + n];
    full[k] = (c >= 2);
  }
  __syncthreads();
  for (int s = threadIdx.x; s < SS; s += blockDim.x)
    mask[b * SS + s] = st[s] | uany[s % NUSR] | full[s / NUSR];
}

// ---------------------------------------------------------------------------
// WMMA GEMM: C[M,N] = A[M,K](f16) @ Wt[N,K](f16,transposed) + bias
// One wave computes a 16x(NTW*16)=16x128 output block with 8 accumulators:
// the A fragment is loaded once per K-step and reused by 8 independent WMMAs.
// Requires Ncols % 128 == 0 (true for 128 and 2048 here).
// ---------------------------------------------------------------------------
__global__ void wmma_gemm(const _Float16* __restrict__ A,
                          const _Float16* __restrict__ Wt,
                          const float* __restrict__ bias,
                          const float* __restrict__ resid,
                          float* __restrict__ outF,
                          _Float16* __restrict__ outH,
                          int M, int Ncols, int Kd, int relu) {
  const int wave  = threadIdx.x >> 5;
  const int lane  = threadIdx.x & 31;
  const int mtile = blockIdx.x * (blockDim.x >> 5) + wave;
  if (mtile * 16 >= M) return;
  const int n0 = blockIdx.y * (NTW * 16);
  const int lm = lane & 15;
  const int hi = lane >> 4;

  const _Float16* arow  = A  + (size_t)(mtile * 16 + lm) * Kd;
  const _Float16* bbase = Wt + (size_t)(n0 + lm) * Kd;

  const float8 z = {0.f, 0.f, 0.f, 0.f, 0.f, 0.f, 0.f, 0.f};
  float8 acc[NTW];
#pragma unroll
  for (int t = 0; t < NTW; ++t) acc[t] = z;

  for (int k0 = 0; k0 < Kd; k0 += 32) {
    union { half16 h; unsigned u[8]; } a;
#pragma unroll
    for (int v = 0; v < 8; ++v) {
      const int ka = k0 + ((v >> 2) << 4) + (hi << 3) + ((v & 3) << 1);  // A layout
      a.u[v] = *(const unsigned*)(arow + ka);
    }
#pragma unroll
    for (int t = 0; t < NTW; ++t) {
      union { half16 h; unsigned u[8]; } b;
      const _Float16* brow = bbase + (size_t)t * 16 * Kd;
#pragma unroll
      for (int v = 0; v < 8; ++v) {
        const int kb = k0 + ((v >> 2) << 4) + (hi << 3) + ((v & 3) << 1);  // B layout
        b.u[v] = *(const unsigned*)(brow + kb);
      }
      acc[t] = __builtin_amdgcn_wmma_f32_16x16x32_f16(false, a.h, false, b.h,
                                                      (short)0, acc[t], false, false);
    }
  }

#pragma unroll
  for (int t = 0; t < NTW; ++t) {
    const int n = n0 + t * 16 + lm;
    const float bval = bias ? bias[n] : 0.f;
#pragma unroll
    for (int v = 0; v < 8; ++v) {
      const size_t m = (size_t)(mtile * 16 + v + 8 * hi);
      float x = acc[t][v] + bval;
      if (resid) x += resid[m * Ncols + n];
      if (relu)  x = x > 0.f ? x : 0.f;
      if (outF)  outF[m * Ncols + n] = x;
      if (outH)  outH[m * Ncols + n] = (_Float16)x;
    }
  }
}

// ---------------------------------------------------------------------------
// Attention: one wave per (batch*head, 16-query tile).
// scores via WMMA (HD=16 zero-padded to K=32), softmax in LDS, P@V via WMMA.
// ---------------------------------------------------------------------------
__global__ void attn_kernel(const _Float16* __restrict__ Q,
                            const _Float16* __restrict__ Kh,
                            const _Float16* __restrict__ V,
                            _Float16* __restrict__ O) {
  __shared__ _Float16 sP[16 * SPAD];           // 32 KB
  const int qt = blockIdx.x;
  const int bh = blockIdx.y;
  const int bb = bh / NHD;
  const int hh = bh % NHD;
  const int lane = threadIdx.x & 31;
  const int lm = lane & 15;
  const int hi = lane >> 4;
  const float scale = 0.25f;                   // 1/sqrt(HD)

  // Q tile in A layout (k >= HD padded with zeros)
  half16 qa;
  const int qrow = qt * 16 + lm;
#pragma unroll
  for (int v = 0; v < 8; ++v) {
    const int ka = ((v >> 2) << 4) + (hi << 3) + ((v & 3) << 1);
    _Float16 e0 = (_Float16)0.f, e1 = (_Float16)0.f;
    if (ka < HDD && qrow < SS) {
      const size_t base = ((size_t)(bb * SS + qrow)) * HH + hh * HDD + ka;
      e0 = Q[base]; e1 = Q[base + 1];
    }
    qa[2 * v] = e0; qa[2 * v + 1] = e1;
  }

  // stage 1: scores -> LDS (OOB keys = -inf); cover full padded width
  for (int kt = 0; kt < SPAD / 16; ++kt) {
    half16 kb;
    const int key = kt * 16 + lm;
#pragma unroll
    for (int v = 0; v < 8; ++v) {
      const int kk = ((v >> 2) << 4) + (hi << 3) + ((v & 3) << 1);
      _Float16 e0 = (_Float16)0.f, e1 = (_Float16)0.f;
      if (kk < HDD && key < SS) {
        const size_t base = ((size_t)(bb * SS + key)) * HH + hh * HDD + kk;
        e0 = Kh[base]; e1 = Kh[base + 1];
      }
      kb[2 * v] = e0; kb[2 * v + 1] = e1;
    }
    float8 acc = {0.f, 0.f, 0.f, 0.f, 0.f, 0.f, 0.f, 0.f};
    acc = __builtin_amdgcn_wmma_f32_16x16x32_f16(false, qa, false, kb,
                                                 (short)0, acc, false, false);
#pragma unroll
    for (int v = 0; v < 8; ++v) {
      const int m = v + 8 * hi;
      const float s = (key < SS) ? acc[v] * scale : -__builtin_inff();
      sP[m * SPAD + kt * 16 + lm] = (_Float16)s;
    }
  }
  __syncthreads();

  // stage 2: softmax per row; two lanes per row (lane lm and lm+16)
  {
    _Float16* rowp = sP + lm * SPAD + hi * (SPAD / 2);
    float mx = -__builtin_inff();
    for (int c = 0; c < SPAD / 2; ++c) mx = fmaxf(mx, (float)rowp[c]);
    mx = fmaxf(mx, __shfl_xor(mx, 16));
    float sum = 0.f;
    for (int c = 0; c < SPAD / 2; ++c) {
      const float e = __expf((float)rowp[c] - mx);
      sum += e;
      rowp[c] = (_Float16)e;
    }
    sum += __shfl_xor(sum, 16);
    const float inv = 1.f / sum;
    for (int c = 0; c < SPAD / 2; ++c) rowp[c] = (_Float16)((float)rowp[c] * inv);
  }
  __syncthreads();

  // stage 3: O tile = P @ V   (K-steps of 32 over padded S)
  float8 oacc = {0.f, 0.f, 0.f, 0.f, 0.f, 0.f, 0.f, 0.f};
  for (int k0 = 0; k0 < SPAD; k0 += 32) {
    half16 pa, vb;
#pragma unroll
    for (int v = 0; v < 8; ++v) {
      const int ka = ((v >> 2) << 4) + (hi << 3) + ((v & 3) << 1);
      pa[2 * v]     = sP[lm * SPAD + k0 + ka];
      pa[2 * v + 1] = sP[lm * SPAD + k0 + ka + 1];
      const int kk = ((v >> 2) << 4) + (hi << 3) + ((v & 3) << 1);
      const int key = k0 + kk;
      _Float16 e0 = (_Float16)0.f, e1 = (_Float16)0.f;
      if (key < SS) {
        const size_t base = ((size_t)(bb * SS + key)) * HH + hh * HDD + lm;
        e0 = V[base];
        if (key + 1 < SS) e1 = V[base + HH];
      }
      vb[2 * v] = e0; vb[2 * v + 1] = e1;
    }
    oacc = __builtin_amdgcn_wmma_f32_16x16x32_f16(false, pa, false, vb,
                                                  (short)0, oacc, false, false);
  }
#pragma unroll
  for (int v = 0; v < 8; ++v) {
    const int m = v + 8 * hi;
    const int s = qt * 16 + m;
    if (s < SS)
      O[((size_t)(bb * SS + s)) * HH + hh * HDD + lm] = (_Float16)oacc[v];
  }
}

// ---------------------------------------------------------------------------
// LayerNorm over H=128, one wave per row (4 elems/lane)
// ---------------------------------------------------------------------------
__global__ void ln_kernel(const float* __restrict__ in, const float* __restrict__ gamma,
                          const float* __restrict__ beta, float* __restrict__ outF,
                          _Float16* __restrict__ outH, int rows) {
  const int row = blockIdx.x * (blockDim.x >> 5) + (threadIdx.x >> 5);
  if (row >= rows) return;
  const int lane = threadIdx.x & 31;
  const float* r = in + (size_t)row * HH;
  float v0 = r[lane], v1 = r[lane + 32], v2 = r[lane + 64], v3 = r[lane + 96];
  float s = v0 + v1 + v2 + v3;
  for (int o = 16; o; o >>= 1) s += __shfl_xor(s, o);
  const float mean = s * (1.f / HH);
  const float d0 = v0 - mean, d1 = v1 - mean, d2 = v2 - mean, d3 = v3 - mean;
  float q = d0 * d0 + d1 * d1 + d2 * d2 + d3 * d3;
  for (int o = 16; o; o >>= 1) q += __shfl_xor(q, o);
  const float inv = rsqrtf(q * (1.f / HH) + 1e-5f);
  float o0 = d0 * inv * gamma[lane]      + beta[lane];
  float o1 = d1 * inv * gamma[lane + 32] + beta[lane + 32];
  float o2 = d2 * inv * gamma[lane + 64] + beta[lane + 64];
  float o3 = d3 * inv * gamma[lane + 96] + beta[lane + 96];
  float* w = outF + (size_t)row * HH;
  _Float16* wh = outH + (size_t)row * HH;
  w[lane] = o0; w[lane + 32] = o1; w[lane + 64] = o2; w[lane + 96] = o3;
  wh[lane] = (_Float16)o0; wh[lane + 32] = (_Float16)o1;
  wh[lane + 64] = (_Float16)o2; wh[lane + 96] = (_Float16)o3;
}

// ---------------------------------------------------------------------------
// Tail kernels
// ---------------------------------------------------------------------------
__global__ void mean_kernel(const float* __restrict__ xF, float* __restrict__ meanF) {
  const int b = blockIdx.x, h = threadIdx.x;     // block = 128
  float s = 0.f;
  for (int i = 0; i < SS; ++i) s += xF[((size_t)(b * SS + i)) * HH + h];
  meanF[b * HH + h] = s * (1.f / SS);
}

__global__ void comb_kernel(const float* __restrict__ meanF, const float* __restrict__ xF,
                            const int* __restrict__ actIdx, const float* __restrict__ W_comb,
                            const float* __restrict__ b_comb, float* __restrict__ gF) {
  const int b = blockIdx.x, h = threadIdx.x;     // block = 128
  const int a = actIdx[b];
  float acc = b_comb[h];
  for (int j = 0; j < HH; ++j) acc += meanF[b * HH + j] * W_comb[(size_t)j * HH + h];
  const float* last = xF + ((size_t)(b * SS + a)) * HH;
  for (int j = 0; j < HH; ++j) acc += last[j] * W_comb[(size_t)(HH + j) * HH + h];
  gF[b * HH + h] = acc;
}

__global__ void score_kernel(const float* __restrict__ decF, const float* __restrict__ gF,
                             float* __restrict__ scoresF) {
  const int idx = blockIdx.x * blockDim.x + threadIdx.x;
  if (idx >= BB * SS) return;
  const int b = idx / SS;
  float acc = 0.f;
  for (int j = 0; j < HH; ++j) acc += decF[(size_t)idx * HH + j] * gF[b * HH + j];
  scoresF[idx] = acc;
}

__global__ void final_kernel(const float* __restrict__ scoresF, const float* __restrict__ W_lin,
                             const float* __restrict__ b_lin, const int* __restrict__ maskB,
                             float* __restrict__ out) {
  const int b = blockIdx.x;
  const int j = blockIdx.y * blockDim.x + threadIdx.x;
  if (j >= SS) return;
  float acc = b_lin[j];
  for (int s = 0; s < SS; ++s) acc += scoresF[b * SS + s] * W_lin[(size_t)s * SS + j];
  out[b * SS + j] = maskB[b * SS + j] ? -__builtin_inff() : acc;
}

// ---------------------------------------------------------------------------
// Host launcher
// ---------------------------------------------------------------------------
extern "C" void kernel_launch(void* const* d_in, const int* in_sizes, int n_in,
                              void* d_out, int out_size, void* d_ws, size_t ws_size,
                              hipStream_t stream) {
  const float* prev_state = (const float*)d_in[0];
  const float* state      = (const float*)d_in[1];
  const float* W_pre      = (const float*)d_in[2];
  const float* b_pre      = (const float*)d_in[3];
  const float* Wq         = (const float*)d_in[4];
  const float* bq         = (const float*)d_in[5];
  const float* Wk         = (const float*)d_in[6];
  const float* bk         = (const float*)d_in[7];
  const float* Wv         = (const float*)d_in[8];
  const float* bv         = (const float*)d_in[9];
  const float* Wo         = (const float*)d_in[10];
  const float* bo         = (const float*)d_in[11];
  const float* ln1_g      = (const float*)d_in[12];
  const float* ln1_b      = (const float*)d_in[13];
  const float* Wff1       = (const float*)d_in[14];
  const float* bff1       = (const float*)d_in[15];
  const float* Wff2       = (const float*)d_in[16];
  const float* bff2       = (const float*)d_in[17];
  const float* ln2_g      = (const float*)d_in[18];
  const float* ln2_b      = (const float*)d_in[19];
  const float* W_comb     = (const float*)d_in[20];
  const float* b_comb     = (const float*)d_in[21];
  const float* W_dec      = (const float*)d_in[22];
  const float* b_dec      = (const float*)d_in[23];
  const float* W_lin      = (const float*)d_in[24];
  const float* b_lin      = (const float*)d_in[25];
  float* out = (float*)d_out;

  char* ws = (char*)d_ws;
  size_t off = 0;
  auto alloc = [&](size_t bytes) -> void* {
    void* p = ws + off;
    off += (bytes + 255) & ~(size_t)255;
    return p;
  };

  float*     xF     = (float*)    alloc((size_t)MM * HH * 4);
  float*     tmpF   = (float*)    alloc((size_t)MM * HH * 4);   // also dec output
  _Float16*  xH     = (_Float16*) alloc((size_t)MM * HH * 2);
  _Float16*  qH     = (_Float16*) alloc((size_t)MM * HH * 2);
  _Float16*  kH     = (_Float16*) alloc((size_t)MM * HH * 2);
  _Float16*  vH     = (_Float16*) alloc((size_t)MM * HH * 2);
  _Float16*  attnH  = (_Float16*) alloc((size_t)MM * HH * 2);
  _Float16*  ffH    = (_Float16*) alloc((size_t)MM * FFD * 2);
  _Float16*  wtA    = (_Float16*) alloc((size_t)FFD * HH * 2);  // reused weight buffer
  float*     meanF  = (float*)    alloc((size_t)BB * HH * 4);
  float*     gF     = (float*)    alloc((size_t)BB * HH * 4);
  float*     scoresF= (float*)    alloc((size_t)BB * SS * 4);
  int*       maskB  = (int*)      alloc((size_t)BB * SS * 4);
  int*       actIdx = (int*)      alloc((size_t)BB * 4);

  const int cvtHH = (HH * HH + 255) / 256;
  const int cvtFF = (HH * FFD + 255) / 256;
  // one wave per 16-row m-tile; 4 waves (128 thr) per block; grid.y = N/128 groups
  const int mBlocks = ((MM / 16) + 3) / 4;              // 125
  const dim3 gemmGridH (mBlocks, HH / (NTW * 16));      // (125, 1)
  const dim3 gemmGridFF(mBlocks, FFD / (NTW * 16));     // (125, 16)

  pre_proj<<<(MM * HH + 255) / 256, 256, 0, stream>>>(state, W_pre, b_pre, xF, xH);
  mask_kernel<<<BB, 256, 0, stream>>>(prev_state, state, maskB, actIdx);

  for (int i = 0; i < LL; ++i) {
    const float* Wq_i = Wq + (size_t)i * HH * HH;
    const float* Wk_i = Wk + (size_t)i * HH * HH;
    const float* Wv_i = Wv + (size_t)i * HH * HH;
    const float* Wo_i = Wo + (size_t)i * HH * HH;
    const float* W1_i = Wff1 + (size_t)i * HH * FFD;
    const float* W2_i = Wff2 + (size_t)i * FFD * HH;

    wt_convert<<<cvtHH, 256, 0, stream>>>(Wq_i, wtA, HH, HH);
    wmma_gemm<<<gemmGridH, 128, 0, stream>>>(xH, wtA, bq + i * HH, nullptr, nullptr, qH,
                                             MM, HH, HH, 0);
    wt_convert<<<cvtHH, 256, 0, stream>>>(Wk_i, wtA, HH, HH);
    wmma_gemm<<<gemmGridH, 128, 0, stream>>>(xH, wtA, bk + i * HH, nullptr, nullptr, kH,
                                             MM, HH, HH, 0);
    wt_convert<<<cvtHH, 256, 0, stream>>>(Wv_i, wtA, HH, HH);
    wmma_gemm<<<gemmGridH, 128, 0, stream>>>(xH, wtA, bv + i * HH, nullptr, nullptr, vH,
                                             MM, HH, HH, 0);

    attn_kernel<<<dim3(STILES, BB * NHD), 32, 0, stream>>>(qH, kH, vH, attnH);

    wt_convert<<<cvtHH, 256, 0, stream>>>(Wo_i, wtA, HH, HH);
    wmma_gemm<<<gemmGridH, 128, 0, stream>>>(attnH, wtA, bo + i * HH, xF, tmpF, nullptr,
                                             MM, HH, HH, 0);
    ln_kernel<<<MM / 8, 256, 0, stream>>>(tmpF, ln1_g + i * HH, ln1_b + i * HH, xF, xH, MM);

    wt_convert<<<cvtFF, 256, 0, stream>>>(W1_i, wtA, HH, FFD);
    wmma_gemm<<<gemmGridFF, 128, 0, stream>>>(xH, wtA, bff1 + i * FFD, nullptr, nullptr, ffH,
                                              MM, FFD, HH, 1);
    wt_convert<<<cvtFF, 256, 0, stream>>>(W2_i, wtA, FFD, HH);
    wmma_gemm<<<gemmGridH, 128, 0, stream>>>(ffH, wtA, bff2 + i * HH, xF, tmpF, nullptr,
                                             MM, HH, FFD, 0);
    ln_kernel<<<MM / 8, 256, 0, stream>>>(tmpF, ln2_g + i * HH, ln2_b + i * HH, xF, xH, MM);
  }

  mean_kernel<<<BB, HH, 0, stream>>>(xF, meanF);
  comb_kernel<<<BB, HH, 0, stream>>>(meanF, xF, actIdx, W_comb, b_comb, gF);

  wt_convert<<<cvtHH, 256, 0, stream>>>(W_dec, wtA, HH, HH);
  wmma_gemm<<<gemmGridH, 128, 0, stream>>>(xH, wtA, b_dec, nullptr, tmpF, nullptr,
                                           MM, HH, HH, 0);

  score_kernel<<<(MM + 255) / 256, 256, 0, stream>>>(tmpF, gF, scoresF);
  final_kernel<<<dim3(BB, (SS + 255) / 256), 256, 0, stream>>>(scoresF, W_lin, b_lin,
                                                               maskB, out);
}
